// Decoder_68367289418033
// MI455X (gfx1250) — matrix operand order repaired and checked by reading, not compiled, not run on hardware
//
#include <hip/hip_runtime.h>
#include <stdint.h>
#include <stddef.h>

// ---------------------------------------------------------------------------
// CDNA5 / gfx1250, wave32.  bf16 WMMA (16x16x32) f32-accumulate, async-LDS
// staging with double buffering, conv+bias folded into the WMMA K-dim.
// ---------------------------------------------------------------------------

typedef __attribute__((ext_vector_type(16))) __bf16 v16bf;
typedef __attribute__((ext_vector_type(2)))  __bf16 v2bf;
typedef __attribute__((ext_vector_type(8)))  float  v8f;
typedef __attribute__((ext_vector_type(4)))  int    v4i;

#define AS1 __attribute__((address_space(1)))
#define AS3 __attribute__((address_space(3)))

#if defined(__AMDGCN__)
#  if __has_builtin(__builtin_amdgcn_global_load_async_to_lds_b128)
#    define USE_ASYNC 1
#  endif
#  if __has_builtin(__builtin_amdgcn_cvt_pk_bf16_f32)
#    define HAVE_PKCVT 1
#  endif
#endif
#ifndef USE_ASYNC
#  define USE_ASYNC 0
#endif
#ifndef HAVE_PKCVT
#  define HAVE_PKCVT 0
#endif

enum { BZ = 64, LL = 2048, DD = 512, UU = 512, AADIM = 512, VV = 32000, EE = 256 };

// ---------------- conversion helpers (native bf16) -------------------------
__device__ __forceinline__ unsigned short bfbits(float f) {
  union { __bf16 h; unsigned short u; } c;
  c.h = (__bf16)f;
  return c.u;
}

__device__ __forceinline__ v2bf pk2(float a, float b) {
#if HAVE_PKCVT
  union { decltype(__builtin_amdgcn_cvt_pk_bf16_f32(0.f, 0.f)) r; v2bf v; } u;
  u.r = __builtin_amdgcn_cvt_pk_bf16_f32(a, b);
  return u.v;
#else
  v2bf r;
  r[0] = (__bf16)a;
  r[1] = (__bf16)b;
  return r;
#endif
}

__device__ __forceinline__ void cvt8(v2bf* d, const float* p) {
  const float4 x0 = *(const float4*)(p);
  const float4 x1 = *(const float4*)(p + 4);
  d[0] = pk2(x0.x, x0.y);
  d[1] = pk2(x0.z, x0.w);
  d[2] = pk2(x1.x, x1.y);
  d[3] = pk2(x1.z, x1.w);
}

__device__ __forceinline__ float fast_tanh(float x) {
  const float ex = __expf(2.0f * x);
  return 1.0f - 2.0f / (ex + 1.0f);
}
__device__ __forceinline__ float fast_sigmoid(float x) {
  return 1.0f / (1.0f + __expf(-x));
}

union Frag16 {
  v16bf v;
  __bf16 e[16];
  v2bf p2[8];
  unsigned short h[16];
  uint4 q[2];
};

template <int N>
__device__ __forceinline__ void wait_async() {
#if USE_ASYNC
#  if __has_builtin(__builtin_amdgcn_s_wait_asynccnt)
  __builtin_amdgcn_s_wait_asynccnt((unsigned short)N);
#  else
  asm volatile("s_wait_asynccnt %0" ::"i"(N) : "memory");
#  endif
#endif
}

// ---------------------------------------------------------------------------
// Generic small GEMM:  D = act( A(MxK) @ B(KxN) + X + Y + bias_row )
// fp32 in/out, bf16 WMMA.  M%16==0, N%128==0, K%32==0.
// One 16x16 tile per wave, 8 waves per block.
// A-frag (16x32 bf16): lane m=lane&15; lanes<16 hold K 0..15,
//                      lanes>=16 K 16..31.
// B-frag (32x16 bf16): lane n=lane&15; lanes<16 K 0..15, lanes>=16 K 16..31.
// C-frag (16x16 f32):  reg r, lane: m = r + 8*(lane>>4), n = lane&15.
// ---------------------------------------------------------------------------
__global__ __launch_bounds__(256) void gemm_bf16_kernel(
    const float* __restrict__ A, const float* __restrict__ B,
    float* __restrict__ D,
    const float* __restrict__ X, const float* __restrict__ Y,
    const float* __restrict__ bias,
    int M, int N, int K, int act)
{
  const int lane = threadIdx.x & 31;
  const int wave = threadIdx.x >> 5;
  const int n0 = blockIdx.x * 128 + wave * 16;
  const int m0 = blockIdx.y * 16;
  if (n0 >= N || m0 >= M) return;

  const int m    = lane & 15;
  const int nn   = lane & 15;
  const int aoff = (lane < 16) ? 0 : 16;
  const int boff = (lane < 16) ? 0 : 16;

  v8f c = {0.f, 0.f, 0.f, 0.f, 0.f, 0.f, 0.f, 0.f};

  for (int k0 = 0; k0 < K; k0 += 32) {
    Frag16 a, b;
    const float* ap = A + (size_t)(m0 + m) * K + k0 + aoff;
    cvt8(a.p2, ap);
    cvt8(a.p2 + 4, ap + 8);

    const float* bp = B + (size_t)(k0 + boff) * N + n0 + nn;
#pragma unroll
    for (int j = 0; j < 8; ++j)
      b.p2[j] = pk2(bp[(size_t)(2 * j) * N], bp[(size_t)(2 * j + 1) * N]);

    c = __builtin_amdgcn_wmma_f32_16x16x32_bf16(false, a.v, false, b.v,
                                                (short)0, c, false, false);
  }

  const int gn = n0 + nn;
  const float bv = bias ? bias[gn] : 0.0f;
#pragma unroll
  for (int r = 0; r < 8; ++r) {
    const int gm = m0 + r + ((lane >> 4) << 3);
    const size_t idx = (size_t)gm * N + gn;
    float v = c[r] + bv;
    if (X) v += X[idx];
    if (Y) v += Y[idx];
    if (act == 1)      v = fast_sigmoid(v);
    else if (act == 2) v = fast_tanh(v);
    D[idx] = v;
  }
}

// ---------------------------------------------------------------------------
// QU[w][n] = sum_k Qk[w,0,k] * Uf[k,n]   (7 x 512): conv kernel folded into Uf
// ---------------------------------------------------------------------------
__global__ void qu_kernel(const float* __restrict__ Qk,
                          const float* __restrict__ Uf,
                          float* __restrict__ QU)
{
  const int i = blockIdx.x * blockDim.x + threadIdx.x;
  if (i >= 7 * 512) return;
  const int w = i / 512, n = i % 512;
  float s = 0.0f;
#pragma unroll 8
  for (int k = 0; k < 64; ++k) s += Qk[w * 64 + k] * Uf[k * 512 + n];
  QU[i] = s;
}

// ---------------------------------------------------------------------------
// Ua (K x N fp32, row-major)  ->  UaT (N x K bf16, row-major)
// ---------------------------------------------------------------------------
__global__ __launch_bounds__(256) void cvt_transpose_kernel(
    const float* __restrict__ src, unsigned short* __restrict__ dst,
    int K, int N)
{
  __shared__ float tile[16][17];
  const int kb = blockIdx.x * 16, nb = blockIdx.y * 16;
  const int tx = threadIdx.x & 15, ty = threadIdx.x >> 4;
  tile[ty][tx] = src[(size_t)(kb + ty) * N + nb + tx];
  __syncthreads();
  dst[(size_t)(nb + ty) * K + kb + tx] = bfbits(tile[tx][ty]);
}

// ---------------------------------------------------------------------------
// Fused attention scores.
//   e[b,l,a] = (features@Ua)[b,l,a] + hWa[b,a] + conv(Bmat,QU)[b,l,a]
//   S[b,l]   = sum_a tanh(e) * Va[a]
// conv + hWa folded into WMMA: K' = 544.  A_aug[l,512+w]=Bw[l+w-3] (w<7),
// A_aug[l,519]=1, rest 0;  B_aug[512+w,n]=QU[w,n], B_aug[519,n]=hWa[b,n].
// B tiles (bf16, pre-transposed UaT) staged to LDS with async-DMA, double
// buffered; aug columns written directly to LDS.
// ---------------------------------------------------------------------------
#define AT_ROWS 128
#define BPITCH 552  // 544 K-columns + 8 pad (row = 1104 B, 16B aligned)

__global__ __launch_bounds__(256) void attn_score_kernel(
    const float* __restrict__ feats,          // (64, 2048, 512)
    const float* __restrict__ Bmat,           // (64, 2048)
    const unsigned short* __restrict__ UaT,   // (512 n, 512 k) bf16
    const float* __restrict__ hWa,            // (64, 512)
    const float* __restrict__ QU,             // (7, 512)
    const float* __restrict__ Va,             // (512)
    float* __restrict__ S)                    // (64, 2048)
{
  const int b    = blockIdx.y;
  const int l0   = blockIdx.x * AT_ROWS;
  const int tid  = threadIdx.x;
  const int wave = tid >> 5;
  const int lane = tid & 31;

  __shared__ unsigned short ldsB[2][16][BPITCH];
  __shared__ float ldsVa[512];
  __shared__ float ldsBw[AT_ROWS + 6];

  for (int i = tid; i < 512; i += 256) ldsVa[i] = Va[i];
  for (int i = tid; i < AT_ROWS + 6; i += 256) {
    const int l = l0 + i - 3;                 // SAME padding, width 7
    ldsBw[i] = (l >= 0 && l < LL) ? Bmat[b * LL + l] : 0.0f;
  }

  // ---- stage tile (K columns 0..511 of UaT rows n0..n0+15) ----
  auto stage_tile = [&](unsigned short(*buf)[BPITCH], int n0) {
    for (int c = tid; c < 1024; c += 256) {          // 1024 x 16B chunks
      const int j  = c >> 6;
      const int ko = (c & 63) << 3;                  // ushort index
      const unsigned short* src = UaT + (size_t)(n0 + j) * 512 + ko;
#if USE_ASYNC
      __builtin_amdgcn_global_load_async_to_lds_b128(
          (AS1 v4i*)(uintptr_t)src, (AS3 v4i*)&buf[j][ko], 0, 0);
#else
      *(uint4*)&buf[j][ko] = *(const uint4*)src;
#endif
    }
  };
  auto stage_aug = [&](unsigned short(*buf)[BPITCH], int n0) {
    for (int i = tid; i < 16 * 32; i += 256) {
      const int n = i >> 5, k = i & 31;
      unsigned short v = 0;
      if (k < 7)       v = bfbits(QU[k * 512 + n0 + n]);
      else if (k == 7) v = bfbits(hWa[b * 512 + n0 + n]);
      buf[n][512 + k] = v;
    }
  };

  stage_tile(ldsB[0], 0);
  __syncthreads();   // consts (Va, Bw) visible to all waves

  // ---- A fragments: 16 rows x 512 K of features, bf16, in VGPRs ----
  Frag16 afr[16];
  {
    const int mr = lane & 15;
    const float* rp = feats + ((size_t)b * LL + l0 + wave * 16 + mr) * DD
                    + ((lane < 16) ? 0 : 16);
#pragma unroll
    for (int kk = 0; kk < 16; ++kk) {
      const float* p = rp + kk * 32;
      cvt8(afr[kk].p2, p);
      cvt8(afr[kk].p2 + 4, p + 8);
    }
  }
  // ---- augmented A fragment (K 512..543): conv taps + 1.0 column ----
  Frag16 aug;
  {
    const int mr = lane & 15;
    if (lane < 16) {
#pragma unroll
      for (int w = 0; w < 7; ++w) aug.e[w] = (__bf16)ldsBw[wave * 16 + mr + w];
      aug.e[7] = (__bf16)1.0f;
#pragma unroll
      for (int j = 8; j < 16; ++j) aug.e[j] = (__bf16)0.0f;
    } else {
#pragma unroll
      for (int j = 0; j < 16; ++j) aug.e[j] = (__bf16)0.0f;
    }
  }

  float sacc[8] = {0.f, 0.f, 0.f, 0.f, 0.f, 0.f, 0.f, 0.f};

  for (int t = 0; t < 32; ++t) {
    const int cur = t & 1;
    if (t + 1 < 32) stage_tile(ldsB[cur ^ 1], (t + 1) * 16);
    stage_aug(ldsB[cur], t * 16);
    if (t < 31) wait_async<4>();   // tile t landed; tile t+1 in flight
    else        wait_async<0>();
    __syncthreads();

    v8f c = {0.f, 0.f, 0.f, 0.f, 0.f, 0.f, 0.f, 0.f};
    const int n = lane & 15;
    const int koffs = (lane < 16) ? 0 : 16;
#pragma unroll
    for (int kk = 0; kk < 17; ++kk) {
      Frag16 bf;
      const uint4* p = (const uint4*)&ldsB[cur][n][kk * 32 + koffs];
      bf.q[0] = p[0];
      bf.q[1] = p[1];
      c = __builtin_amdgcn_wmma_f32_16x16x32_bf16(
          false, (kk < 16) ? afr[kk].v : aug.v, false, bf.v,
          (short)0, c, false, false);
    }

    const float va = ldsVa[t * 16 + n];
#pragma unroll
    for (int r = 0; r < 8; ++r) sacc[r] += fast_tanh(c[r]) * va;
    __syncthreads();
  }

  // Reduce 16 column-lanes of each half-wave -> one score per row.
#pragma unroll
  for (int r = 0; r < 8; ++r) {
    float v = sacc[r];
    v += __shfl_xor(v, 1, 32);
    v += __shfl_xor(v, 2, 32);
    v += __shfl_xor(v, 4, 32);
    v += __shfl_xor(v, 8, 32);
    if ((lane & 15) == 0) {
      const int mrow = wave * 16 + r + ((lane >> 4) << 3);
      S[(size_t)b * LL + l0 + mrow] = v;
    }
  }
}

// ---------------------------------------------------------------------------
// Softmax over L per batch row -> attention weights.
// ---------------------------------------------------------------------------
__global__ __launch_bounds__(256) void softmax_L_kernel(
    const float* __restrict__ S, float* __restrict__ attn)
{
  const int b = blockIdx.x;
  const int tid = threadIdx.x;
  __shared__ float red[256];

  float mx = -3.4e38f;
  for (int i = tid; i < LL; i += 256) mx = fmaxf(mx, S[b * LL + i]);
  red[tid] = mx; __syncthreads();
  for (int s = 128; s > 0; s >>= 1) {
    if (tid < s) red[tid] = fmaxf(red[tid], red[tid + s]);
    __syncthreads();
  }
  mx = red[0]; __syncthreads();

  float sum = 0.0f;
  for (int i = tid; i < LL; i += 256) sum += __expf(S[b * LL + i] - mx);
  red[tid] = sum; __syncthreads();
  for (int s = 128; s > 0; s >>= 1) {
    if (tid < s) red[tid] += red[tid + s];
    __syncthreads();
  }
  const float inv = 1.0f / red[0];
  for (int i = tid; i < LL; i += 256)
    attn[b * LL + i] = __expf(S[b * LL + i] - mx) * inv;
}

// ---------------------------------------------------------------------------
// context[b, d] = sum_l attn[b,l] * features[b,l,d]   (pure bandwidth)
// ---------------------------------------------------------------------------
__global__ __launch_bounds__(256) void context_kernel(
    const float* __restrict__ attn, const float* __restrict__ feats,
    float* __restrict__ ctx)
{
  const int b = blockIdx.x;
  const int tid = threadIdx.x;
  float a0 = 0.0f, a1 = 0.0f;
  const float* f = feats + (size_t)b * LL * DD;
  for (int l = 0; l < LL; ++l) {
    const float a = attn[b * LL + l];
    a0 += a * f[(size_t)l * DD + tid];
    a1 += a * f[(size_t)l * DD + 256 + tid];
  }
  ctx[b * DD + tid]       = a0;
  ctx[b * DD + 256 + tid] = a1;
}

// ---------------------------------------------------------------------------
// Elementwise helpers
// ---------------------------------------------------------------------------
__global__ void ewmul_kernel(const float* __restrict__ a, const float* __restrict__ b,
                             float* __restrict__ o, int n)
{
  const int i = blockIdx.x * blockDim.x + threadIdx.x;
  if (i < n) o[i] = a[i] * b[i];
}

__global__ void gru_gate_kernel(const float* __restrict__ z, const float* __restrict__ h,
                                const float* __restrict__ hc, float* __restrict__ ht, int n)
{
  const int i = blockIdx.x * blockDim.x + threadIdx.x;
  if (i < n) ht[i] = (1.0f - z[i]) * h[i] + z[i] * hc[i];
}

// ---------------------------------------------------------------------------
// Row softmax over N (=32000) -> final probabilities.
// ---------------------------------------------------------------------------
__global__ __launch_bounds__(256) void softmax_row_kernel(
    const float* __restrict__ X, float* __restrict__ Y, int N)
{
  const int row = blockIdx.x;
  const int tid = threadIdx.x;
  __shared__ float red[256];
  const float* x = X + (size_t)row * N;
  float* y = Y + (size_t)row * N;

  float mx = -3.4e38f;
  for (int i = tid; i < N; i += 256) mx = fmaxf(mx, x[i]);
  red[tid] = mx; __syncthreads();
  for (int s = 128; s > 0; s >>= 1) {
    if (tid < s) red[tid] = fmaxf(red[tid], red[tid + s]);
    __syncthreads();
  }
  mx = red[0]; __syncthreads();

  float sum = 0.0f;
  for (int i = tid; i < N; i += 256) sum += __expf(x[i] - mx);
  red[tid] = sum; __syncthreads();
  for (int s = 128; s > 0; s >>= 1) {
    if (tid < s) red[tid] += red[tid + s];
    __syncthreads();
  }
  const float inv = 1.0f / red[0];
  for (int i = tid; i < N; i += 256) y[i] = __expf(x[i] - mx) * inv;
}

// ---------------------------------------------------------------------------
// Host-side orchestration
// ---------------------------------------------------------------------------
extern "C" void kernel_launch(void* const* d_in, const int* in_sizes, int n_in,
                              void* d_out, int out_size, void* d_ws, size_t ws_size,
                              hipStream_t stream)
{
  (void)in_sizes; (void)n_in; (void)out_size; (void)ws_size;

  const float* prev_target = (const float*)d_in[0];   // (64, 32000)
  const float* prev_hidden = (const float*)d_in[1];   // (64, 512)
  const float* features    = (const float*)d_in[2];   // (64, 2048, 512)
  const float* Bmat        = (const float*)d_in[3];   // (64, 2048, 1)
  const float* Wa          = (const float*)d_in[4];   // (512, 512)
  const float* Ua          = (const float*)d_in[5];   // (512, 512)
  const float* Va          = (const float*)d_in[6];   // (512, 1)
  const float* Uf          = (const float*)d_in[7];   // (64, 512)
  const float* Qk          = (const float*)d_in[8];   // (7, 1, 64)
  const float* We          = (const float*)d_in[9];   // (32000, 256)
  const float* Wccz        = (const float*)d_in[10];  // (512, 512)
  const float* Wyz         = (const float*)d_in[11];  // (256, 512)
  const float* Uhz         = (const float*)d_in[12];  // (512, 512)
  const float* Wyr         = (const float*)d_in[13];  // (256, 512)
  const float* Uhr         = (const float*)d_in[14];  // (512, 512)
  const float* Ccr         = (const float*)d_in[15];  // (512, 512)
  const float* Wyh         = (const float*)d_in[16];  // (256, 512)
  const float* b_wyh       = (const float*)d_in[17];  // (512)
  const float* Urh         = (const float*)d_in[18];  // (512, 512)
  const float* b_urh       = (const float*)d_in[19];  // (512)
  const float* Wo          = (const float*)d_in[20];  // (256, 32000)
  const float* Wh          = (const float*)d_in[21];  // (512, 256)
  const float* Wc          = (const float*)d_in[22];  // (512, 256)

  float* out = (float*)d_out;
  const size_t OFF_HT   = (size_t)BZ * VV;            // 2,048,000
  const size_t OFF_ATTN = OFF_HT + (size_t)BZ * UU;   // 2,080,768

  float* wsf = (float*)d_ws;
  size_t off = 0;
  auto take = [&](size_t n) { float* p = wsf + off; off += (n + 63) & ~(size_t)63; return p; };
  float* w_QU     = take(7 * 512);
  float* w_hWa    = take((size_t)BZ * AADIM);
  float* w_UaT_f  = take((size_t)512 * 512 / 2);      // bf16 transposed Ua
  float* w_S      = take((size_t)BZ * LL);
  float* w_ctx    = take((size_t)BZ * DD);
  float* w_Ey     = take((size_t)BZ * EE);
  float* w_ccz    = take((size_t)BZ * UU);
  float* w_t1     = take((size_t)BZ * UU);
  float* w_t2     = take((size_t)BZ * UU);
  float* w_t3     = take((size_t)BZ * UU);
  float* w_t4     = take((size_t)BZ * UU);
  float* w_zt     = take((size_t)BZ * UU);
  float* w_rt     = take((size_t)BZ * UU);
  float* w_rh     = take((size_t)BZ * UU);
  float* w_hc     = take((size_t)BZ * UU);
  float* w_o1     = take((size_t)BZ * EE);
  float* w_opre   = take((size_t)BZ * EE);
  float* w_logits = take((size_t)BZ * VV);
  unsigned short* w_UaT = (unsigned short*)w_UaT_f;

  // 1) Fold conv kernel into Uf:  QU = Qk @ Uf  (7 x 512)
  qu_kernel<<<14, 256, 0, stream>>>(Qk, Uf, w_QU);

  // 2) hWa = prev_hidden @ Wa  (64 x 512)
  gemm_bf16_kernel<<<dim3(4, 4), 256, 0, stream>>>(
      prev_hidden, Wa, w_hWa, nullptr, nullptr, nullptr, BZ, AADIM, UU, 0);

  // 3) Ua -> bf16 transposed copy (512 x 512)
  cvt_transpose_kernel<<<dim3(32, 32), 256, 0, stream>>>(Ua, w_UaT, 512, 512);

  // 4) Fused attention energies + tanh + Va dot -> S (64 x 2048)
  attn_score_kernel<<<dim3(LL / AT_ROWS, BZ), 256, 0, stream>>>(
      features, Bmat, w_UaT, w_hWa, w_QU, Va, w_S);

  // 5) Softmax over L -> attn (third output)
  softmax_L_kernel<<<BZ, 256, 0, stream>>>(w_S, out + OFF_ATTN);

  // 6) context = sum_l attn * features  (64 x 512)
  context_kernel<<<BZ, 256, 0, stream>>>(out + OFF_ATTN, features, w_ctx);

  // 7) Ey = prev_target @ We  (64 x 256, K = 32000)
  gemm_bf16_kernel<<<dim3(2, 4), 256, 0, stream>>>(
      prev_target, We, w_Ey, nullptr, nullptr, nullptr, BZ, EE, VV, 0);

  // 8) GRU gates
  gemm_bf16_kernel<<<dim3(4, 4), 256, 0, stream>>>(
      w_ctx, Wccz, w_ccz, nullptr, nullptr, nullptr, BZ, UU, DD, 0);
  gemm_bf16_kernel<<<dim3(4, 4), 256, 0, stream>>>(
      w_Ey, Wyz, w_t1, nullptr, nullptr, nullptr, BZ, UU, EE, 0);
  gemm_bf16_kernel<<<dim3(4, 4), 256, 0, stream>>>(       // zt = sigmoid(h@Uhz + t1 + ccz)
      prev_hidden, Uhz, w_zt, w_t1, w_ccz, nullptr, BZ, UU, UU, 1);
  gemm_bf16_kernel<<<dim3(4, 4), 256, 0, stream>>>(
      w_Ey, Wyr, w_t2, nullptr, nullptr, nullptr, BZ, UU, EE, 0);
  gemm_bf16_kernel<<<dim3(4, 4), 256, 0, stream>>>(       // t3 = ctx@Ccr + t2
      w_ctx, Ccr, w_t3, w_t2, nullptr, nullptr, BZ, UU, DD, 0);
  gemm_bf16_kernel<<<dim3(4, 4), 256, 0, stream>>>(       // rt = sigmoid(h@Uhr + t3)
      prev_hidden, Uhr, w_rt, w_t3, nullptr, nullptr, BZ, UU, UU, 1);
  ewmul_kernel<<<128, 256, 0, stream>>>(w_rt, prev_hidden, w_rh, BZ * UU);
  gemm_bf16_kernel<<<dim3(4, 4), 256, 0, stream>>>(       // t4 = Ey@Wyh + ccz + b_wyh
      w_Ey, Wyh, w_t4, w_ccz, nullptr, b_wyh, BZ, UU, EE, 0);
  gemm_bf16_kernel<<<dim3(4, 4), 256, 0, stream>>>(       // hc = tanh(rh@Urh + t4 + b_urh)
      w_rh, Urh, w_hc, w_t4, nullptr, b_urh, BZ, UU, UU, 2);
  gru_gate_kernel<<<128, 256, 0, stream>>>(w_zt, prev_hidden, w_hc, out + OFF_HT, BZ * UU);

  // 9) Output projection + vocabulary softmax
  gemm_bf16_kernel<<<dim3(2, 4), 256, 0, stream>>>(       // o1 = ht@Wh + Ey
      out + OFF_HT, Wh, w_o1, w_Ey, nullptr, nullptr, BZ, EE, UU, 0);
  gemm_bf16_kernel<<<dim3(2, 4), 256, 0, stream>>>(       // opre = ctx@Wc + o1
      w_ctx, Wc, w_opre, w_o1, nullptr, nullptr, BZ, EE, DD, 0);
  gemm_bf16_kernel<<<dim3(VV / 128, 4), 256, 0, stream>>>(// logits = opre @ Wo
      w_opre, Wo, w_logits, nullptr, nullptr, nullptr, BZ, VV, EE, 0);
  softmax_row_kernel<<<BZ, 256, 0, stream>>>(w_logits, out, VV);
}
